// VQCAE_30124900614795
// MI455X (gfx1250) — compile-verified, matrix operand order checked by reading, not run on hardware
//
#include <hip/hip_runtime.h>
#include <hip/hip_bf16.h>
#include <math.h>

typedef __attribute__((ext_vector_type(16))) _Float16 v16h;
typedef __attribute__((ext_vector_type(8)))  _Float16 v8h;
typedef __attribute__((ext_vector_type(8)))  float    v8f;

// ---------------------------------------------------------------------------
// Implicit-GEMM 3x3 conv via WMMA f32_16x16x32_f16 (wave32).
//   Tap-major K ordering: k' = tap*Cinp + ci, Cinp = ceil32(Cin) (pow2 here).
//   Activations are stored as f16 NHWC between layers: a lane's B fragment is
//   two contiguous 8-half runs -> two global b128 loads, no cvt in the loop.
//   A panel staged in LDS in fragment order -> two 16B ds loads per fragment.
//   Wave tile: M=16 out-channels x N=64 positions (4 accumulators, A reused).
//   Block = 8 waves -> 16 co x 512 positions.
//   dil2=1   -> lhs-dilated conv (ConvTranspose2d k3 s2 p1 op1), stride==1
//   wTrans=1 -> weights stored (Cin,Cout,3,3), flipped taps (deconv weights)
//   fuse: 0=none, 1=relu, 2=tanh
//   inNHWC=1: read xh (f16 NHWC); else read xf (f32 NCHW, conv1 only)
//   outNCHW=1: write yf (f32 NCHW, final conv); else write yh (f16 NHWC)
// ---------------------------------------------------------------------------
__global__ __launch_bounds__(256)
void conv_wmma(const float* __restrict__ xf, const _Float16* __restrict__ xh,
               const float* __restrict__ w, const float* __restrict__ bias,
               _Float16* __restrict__ yh, float* __restrict__ yf,
               int Bn, int Cin, int Hi, int Wi, int Cout, int Ho, int Wo,
               int stride, int dil2, int wTrans, int fuse, int inNHWC, int outNCHW)
{
    extern __shared__ char smem[];
    const int Cinp = (Cin + 31) & ~31;           // 32/64/128 (power of two)
    const int nKs  = Cinp >> 5;                  // 1/2/4 K-steps per tap
    const int nksh = (nKs == 1) ? 0 : ((nKs == 2) ? 1 : 2);

    const int tid    = threadIdx.x;
    const int cobase = blockIdx.y * 16;
    const int HiWi   = Hi * Wi;
    const int HoWo   = Ho * Wo;

    // Stage A panel into LDS as f16 in *fragment order*:
    //   half index i = ((tap*nKs + cbi)*32 + lane)*16 + hh
    _Float16* aSh = (_Float16*)smem;
    for (int i = tid; i < 9 * nKs * 512; i += 256) {
        int hh  = i & 15;
        int l   = (i >> 4) & 31;
        int cbi = (i >> 9) & (nKs - 1);
        int tap = i >> (9 + nksh);
        int gg = l >> 4, m = l & 15;
        int koff = hh + ((hh >= 8) ? 8 : 0) + 8 * gg;   // ISA 7.12.2 K pattern
        int ci = (cbi << 5) + koff;
        int kh = tap / 3, kw = tap - kh * 3;
        int co = cobase + m;
        float v = 0.f;
        if (ci < Cin && co < Cout) {
            v = wTrans ? w[((ci * Cout + co) * 3 + (2 - kh)) * 3 + (2 - kw)]
                       : w[((co * Cin + ci) * 3 + kh) * 3 + kw];
        }
        aSh[i] = (_Float16)v;
    }
    __syncthreads();

    const int lane = tid & 31;
    const int wave = tid >> 5;
    const int g  = lane >> 4;
    const int mn = lane & 15;
    const int P  = Bn * HoWo;

    // Per-lane info for the 4 position blocks (N = 64 per wave)
    int  hbA[4], wbA[4];
    const float*    xfA[4];
    const _Float16* xhA[4];
    long ybA[4];
    bool pvA[4];
    #pragma unroll
    for (int nb = 0; nb < 4; ++nb) {
        int p = blockIdx.x * 512 + wave * 64 + nb * 16 + mn;
        bool pv = (p < P);
        if (!pv) p = 0;
        int b  = p / HoWo;
        int rp = p - b * HoWo;
        int ho = rp / Wo;
        int wo = rp - ho * Wo;
        pvA[nb] = pv;
        hbA[nb] = ho * stride - 1;   // pad = 1 everywhere
        wbA[nb] = wo * stride - 1;
        xfA[nb] = xf ? (xf + (size_t)b * Cin * HiWi) : nullptr;
        xhA[nb] = xh ? (xh + (size_t)b * HiWi * Cin) : nullptr;
        ybA[nb] = outNCHW ? ((long)b * Cout * HoWo + rp)
                          : (((long)b * HoWo + rp) * (long)Cout);
    }

    int koffB[16];
    #pragma unroll
    for (int hh = 0; hh < 16; ++hh)
        koffB[hh] = hh + ((hh >= 8) ? 8 : 0) + 8 * g;

    v8f acc[4] = {v8f{}, v8f{}, v8f{}, v8f{}};
    const v8h* aF8 = (const v8h*)smem;

    #pragma unroll
    for (int tap = 0; tap < 9; ++tap) {
        const int kh = tap / 3, kw = tap % 3;        // compile-time constants
        const _Float16* pTh[4];
        const float*    pTf[4];
        bool okT[4];
        #pragma unroll
        for (int nb = 0; nb < 4; ++nb) {
            int dh = hbA[nb] + kh, dw = wbA[nb] + kw;
            int hi, wi; bool ok = pvA[nb];
            if (dil2) {
                ok = ok && dh >= 0 && dw >= 0 && !(dh & 1) && !(dw & 1);
                hi = dh >> 1; wi = dw >> 1;
                ok = ok && hi < Hi && wi < Wi;
            } else {
                hi = dh; wi = dw;
                ok = ok && hi >= 0 && hi < Hi && wi >= 0 && wi < Wi;
            }
            okT[nb] = ok;
            int off = ok ? (hi * Wi + wi) : 0;
            if (inNHWC) pTh[nb] = xhA[nb] + (size_t)off * Cin;
            else        pTf[nb] = xfA[nb] + (size_t)off;
        }

        for (int cbi = 0; cbi < nKs; ++cbi) {
            // A fragment: two 16B LDS loads in fragment order
            int fidx = ((tap * nKs + cbi) * 32 + lane) * 2;
            v8h a0 = aF8[fidx], a1 = aF8[fidx + 1];
            v16h afr;
            #pragma unroll
            for (int j = 0; j < 8; ++j) { afr[j] = a0[j]; afr[8 + j] = a1[j]; }

            v16h bfr[4];
            if (inNHWC) {
                int c0 = (cbi << 5) + 8 * g;         // first contiguous run
                #pragma unroll
                for (int nb = 0; nb < 4; ++nb) {
                    const v8h* q = (const v8h*)(pTh[nb] + c0);
                    v8h h0{}, h1{};
                    if (okT[nb]) { h0 = q[0]; h1 = q[2]; }  // +0..7, +16..23
                    #pragma unroll
                    for (int j = 0; j < 8; ++j) {
                        bfr[nb][j]     = h0[j];
                        bfr[nb][8 + j] = h1[j];
                    }
                }
                if (cbi + 1 < nKs && okT[0])
                    __builtin_prefetch(pTh[0] + c0 + 32, 0, 0);  // speculative
            } else {
                // scalar NCHW f32 gather (conv1 only, Cin=3) with clamp
                #pragma unroll
                for (int hh = 0; hh < 16; ++hh) {
                    int ci  = (cbi << 5) + koffB[hh];
                    int cic = (ci < Cin) ? ci : 0;   // A is zero there anyway
                    int cio = cic * HiWi;
                    #pragma unroll
                    for (int nb = 0; nb < 4; ++nb)
                        bfr[nb][hh] = okT[nb] ? (_Float16)pTf[nb][cio]
                                              : (_Float16)0.f;
                }
            }
            #pragma unroll
            for (int nb = 0; nb < 4; ++nb)
                acc[nb] = __builtin_amdgcn_wmma_f32_16x16x32_f16(
                    false, afr, false, bfr[nb], (short)0, acc[nb], false, false);
        }
    }

    // D layout: lane group g, VGPR r -> row M = r + 8*g, col N = mn
    #pragma unroll
    for (int r = 0; r < 8; ++r) {
        int co = cobase + r + 8 * g;
        if (co < Cout) {
            float bsv = bias[co];
            #pragma unroll
            for (int nb = 0; nb < 4; ++nb) {
                if (pvA[nb]) {
                    float o = acc[nb][r] + bsv;
                    if (fuse == 1)      o = fmaxf(o, 0.f);
                    else if (fuse == 2) o = tanhf(o);
                    if (outNCHW) yf[ybA[nb] + (long)co * HoWo] = o;
                    else         yh[ybA[nb] + co] = (_Float16)o;
                }
            }
        }
    }
}

// ---------------------------------------------------------------------------
// BatchNorm (training mode) on f16 NHWC, two-stage deterministic reduction.
// ---------------------------------------------------------------------------
__global__ __launch_bounds__(256)
void bn_stats1(const _Float16* __restrict__ x, float* __restrict__ part,
               int NP, int C, int csh)
{
    __shared__ float sh[256], sh2[256];
    const int tid = threadIdx.x;
    const int c  = tid & (C - 1);
    const int pl = tid >> csh;          // position lane within block step
    const int PT = 256 >> csh;          // positions covered per block step
    float s = 0.f, s2 = 0.f;
    for (int p = blockIdx.x * PT + pl; p < NP; p += 256 * PT) {
        float v = (float)x[(size_t)p * C + c];
        s += v; s2 += v * v;
    }
    sh[tid] = s; sh2[tid] = s2;
    __syncthreads();
    if (tid < C) {
        for (int k = 1; k < PT; ++k) { s += sh[tid + k * C]; s2 += sh2[tid + k * C]; }
        part[blockIdx.x * C + tid]           = s;
        part[256 * C + blockIdx.x * C + tid] = s2;
    }
}

__global__ __launch_bounds__(256)
void bn_stats2(const float* __restrict__ part, float* __restrict__ mean,
               float* __restrict__ var, int C, int N)
{
    int c = threadIdx.x;
    if (c >= C) return;
    float s = 0.f, s2 = 0.f;
    for (int b = 0; b < 256; ++b) {
        s  += part[b * C + c];
        s2 += part[256 * C + b * C + c];
    }
    float m = s / (float)N;
    mean[c] = m;
    var[c]  = fmaxf(s2 / (float)N - m * m, 0.f);
}

// f16 NHWC normalize + ReLU; optional second store in f32 NCHW (z_e output).
__global__ __launch_bounds__(256)
void bn_apply_relu(const _Float16* __restrict__ x, const float* __restrict__ mean,
                   const float* __restrict__ var, const float* __restrict__ gm,
                   const float* __restrict__ bt, _Float16* __restrict__ y,
                   float* __restrict__ ynchw, int C, int csh, int HW, size_t n)
{
    size_t i = (size_t)blockIdx.x * 256 + threadIdx.x;
    if (i >= n) return;
    int c = (int)(i & (size_t)(C - 1));
    float v = ((float)x[i] - mean[c]) * rsqrtf(var[c] + 1e-5f) * gm[c] + bt[c];
    v = fmaxf(v, 0.f);
    y[i] = (_Float16)v;
    if (ynchw) {
        size_t pos = i >> csh;
        size_t b = pos / (size_t)HW, s = pos - b * (size_t)HW;
        ynchw[((size_t)b * C + c) * HW + s] = v;
    }
}

// ---------------------------------------------------------------------------
// Vector quantization on f16 NHWC z: argmin_k |z-cb_k|^2 via |cb|^2 - 2 z.cb.
// Writes z_q in f16 NHWC (internal) and f32 NCHW (output).
// ---------------------------------------------------------------------------
__global__ __launch_bounds__(256)
void vq_kernel(const _Float16* __restrict__ z, const float* __restrict__ cbk,
               _Float16* __restrict__ zqh, float* __restrict__ zqn, int HW)
{
    __shared__ float zs[16][128];
    __shared__ float bsc[16][16];
    __shared__ int   bix[16][16];
    __shared__ int   fin[16];
    const int t = threadIdx.x;
    const int pblock = blockIdx.x * 16;

    for (int i = t; i < 16 * 128; i += 256) {
        int pi = i >> 7, d = i & 127;
        zs[pi][d] = (float)z[(size_t)(pblock + pi) * 128 + d];   // coalesced
    }
    __syncthreads();

    const int pi = t >> 4, sub = t & 15;
    float best = 3.4e38f; int bidx = 0;
    for (int code = sub; code < 128; code += 16) {
        const float* cr = cbk + code * 128;
        float acc = 0.f;
        #pragma unroll 8
        for (int d = 0; d < 128; ++d) {
            float cv = cr[d];
            acc += cv * cv - 2.f * cv * zs[pi][d];
        }
        if (acc < best) { best = acc; bidx = code; }
    }
    bsc[pi][sub] = best; bix[pi][sub] = bidx;
    __syncthreads();
    if (sub == 0) {
        float bb = bsc[pi][0]; int bc = bix[pi][0];
        for (int k = 1; k < 16; ++k) {
            float v = bsc[pi][k]; int c = bix[pi][k];
            if (v < bb || (v == bb && c < bc)) { bb = v; bc = c; }
        }
        fin[pi] = bc;
    }
    __syncthreads();
    int p = pblock + pi;
    int b = p / HW, s = p - b * HW;
    const float* cr = cbk + fin[pi] * 128;
    for (int c = sub; c < 128; c += 16) {
        float v = cr[c];
        zqh[(size_t)p * 128 + c] = (_Float16)v;
        zqn[((size_t)b * 128 + c) * HW + s] = v;
    }
}

// ---------------------------------------------------------------------------
// Host-side orchestration
// ---------------------------------------------------------------------------
extern "C" void kernel_launch(void* const* d_in, const int* in_sizes, int n_in,
                              void* d_out, int out_size, void* d_ws, size_t ws_size,
                              hipStream_t stream)
{
    (void)in_sizes; (void)n_in; (void)out_size; (void)ws_size;

    const float* x    = (const float*)d_in[0];
    const float* w1   = (const float*)d_in[1];  const float* b1   = (const float*)d_in[2];
    const float* w2   = (const float*)d_in[3];  const float* b2   = (const float*)d_in[4];
    const float* w3   = (const float*)d_in[5];  const float* b3   = (const float*)d_in[6];
    const float* w4   = (const float*)d_in[7];  const float* b4   = (const float*)d_in[8];
    const float* g4   = (const float*)d_in[9];  const float* be4  = (const float*)d_in[10];
    const float* w5   = (const float*)d_in[11]; const float* b5   = (const float*)d_in[12];
    const float* g5   = (const float*)d_in[13]; const float* be5  = (const float*)d_in[14];
    const float* cb   = (const float*)d_in[15];
    const float* dw1  = (const float*)d_in[16]; const float* db1  = (const float*)d_in[17];
    const float* dg1  = (const float*)d_in[18]; const float* dbe1 = (const float*)d_in[19];
    const float* dwt2 = (const float*)d_in[20]; const float* dbt2 = (const float*)d_in[21];
    const float* dg2  = (const float*)d_in[22]; const float* dbe2 = (const float*)d_in[23];
    const float* dwt3 = (const float*)d_in[24]; const float* dbt3 = (const float*)d_in[25];
    const float* dwt4 = (const float*)d_in[26]; const float* dbt4 = (const float*)d_in[27];
    const float* dw5  = (const float*)d_in[28]; const float* db5  = (const float*)d_in[29];

    _Float16* wsh = (_Float16*)d_ws;
    float*    out = (float*)d_out;

    // workspace layout (f16 halves, NHWC internal), 64-half slack per region
    constexpr size_t SLK  = 64;
    constexpr size_t SZ_A = 71303168;  // 64*136*512*16  a1 / u2n / u4
    constexpr size_t SZ_B = 35651584;  // 64*68*256*32   a2 / u3
    constexpr size_t SZ_C = 17825792;  // 64*34*128*64   a3 / u2r
    constexpr size_t SZ_D = 8912896;   // 64*17*64*128
    constexpr size_t O_A = 0;
    constexpr size_t O_B = O_A + SZ_A + SLK;
    constexpr size_t O_C = O_B + SZ_B + SLK;
    constexpr size_t O_D = O_C + SZ_C + SLK;   // a4r / a5r / d1r
    constexpr size_t O_E = O_D + SZ_D + SLK;   // a4n / zqh
    constexpr size_t O_F = O_E + SZ_D + SLK;   // zeh / d1n
    constexpr size_t O_S = O_F + SZ_D + SLK;   // f32 stats (even offset)

    _Float16* a1  = wsh + O_A;  _Float16* a2  = wsh + O_B;  _Float16* a3 = wsh + O_C;
    _Float16* a4r = wsh + O_D;  _Float16* a4n = wsh + O_E;
    _Float16* a5r = wsh + O_D;  _Float16* zeh = wsh + O_F;  // NHWC z_e (internal)
    _Float16* zqh = wsh + O_E;  _Float16* d1r = wsh + O_D;  _Float16* d1n = wsh + O_F;
    _Float16* u2r = wsh + O_C;  _Float16* u2n = wsh + O_A;
    _Float16* u3  = wsh + O_B;  _Float16* u4  = wsh + O_A;
    float* mS   = (float*)(wsh + O_S);
    float* vS   = mS + 128;
    float* part = mS + 256;                    // 2*256*128 floats

    constexpr size_t RECON = 13369344;  // 64*3*136*512
    constexpr size_t ZSZ   = 8912896;
    float* recon = out;
    float* z_e   = out + RECON;          // f32 NCHW output
    float* z_q   = out + RECON + ZSZ;    // f32 NCHW output

    auto conv = [&](const float* inf, const _Float16* inh,
                    const float* wt, const float* bs,
                    _Float16* oh, float* of,
                    int Cin, int Hi, int Wi, int Cout, int Ho, int Wo,
                    int s, int dil2, int tr, int fuse, int inNHWC, int outNCHW) {
        int P = 64 * Ho * Wo;
        dim3 grid((P + 511) / 512, (Cout + 15) / 16);
        int Cinp = (Cin + 31) & ~31;
        size_t lds = (size_t)(9 * Cinp) * 16 * sizeof(_Float16);
        conv_wmma<<<grid, 256, lds, stream>>>(
            inf, inh, wt, bs, oh, of, 64, Cin, Hi, Wi, Cout, Ho, Wo,
            s, dil2, tr, fuse, inNHWC, outNCHW);
    };
    auto bn = [&](const _Float16* raw, const float* gg, const float* bb,
                  _Float16* o, float* onchw, int C, int csh, int HW) {
        int NP = 64 * HW;
        bn_stats1<<<256, 256, 0, stream>>>(raw, part, NP, C, csh);
        bn_stats2<<<1, 256, 0, stream>>>(part, mS, vS, C, NP);
        size_t n = (size_t)NP * C;
        bn_apply_relu<<<(unsigned)((n + 255) / 256), 256, 0, stream>>>(
            raw, mS, vS, gg, bb, o, onchw, C, csh, HW, n);
    };

    // encoder (a1.. all f16 NHWC internal)
    conv(x,    nullptr, w1, b1, a1,  nullptr,  3, 136, 512, 16, 136, 512, 1,0,0, 1, 0,0);
    conv(nullptr, a1,  w2, b2, a2,  nullptr, 16, 136, 512, 32,  68, 256, 2,0,0, 1, 1,0);
    conv(nullptr, a2,  w3, b3, a3,  nullptr, 32,  68, 256, 64,  34, 128, 2,0,0, 1, 1,0);
    conv(nullptr, a3,  w4, b4, a4r, nullptr, 64,  34, 128,128,  17,  64, 2,0,0, 0, 1,0);
    bn(a4r, g4, be4, a4n, nullptr, 128, 7, 17 * 64);
    conv(nullptr, a4n, w5, b5, a5r, nullptr,128,  17,  64,128,  17,  64, 1,0,0, 0, 1,0);
    bn(a5r, g5, be5, zeh, z_e, 128, 7, 17 * 64);       // dual: f16 NHWC + f32 NCHW

    // vector quantization: zeh -> zqh (f16 NHWC) + z_q (f32 NCHW out)
    vq_kernel<<<4352, 256, 0, stream>>>(zeh, cb, zqh, z_q, 17 * 64);

    // decoder
    conv(nullptr, zqh, dw1, db1, d1r, nullptr,128, 17, 64,128, 17, 64, 1,0,0, 0, 1,0);
    bn(d1r, dg1, dbe1, d1n, nullptr, 128, 7, 17 * 64);
    conv(nullptr, d1n, dwt2, dbt2, u2r, nullptr,128, 17, 64, 64, 34,128, 1,1,1, 0, 1,0);
    bn(u2r, dg2, dbe2, u2n, nullptr, 64, 6, 34 * 128);
    conv(nullptr, u2n, dwt3, dbt3, u3, nullptr, 64, 34,128, 32, 68,256, 1,1,1, 1, 1,0);
    conv(nullptr, u3,  dwt4, dbt4, u4, nullptr, 32, 68,256, 16,136,512, 1,1,1, 1, 1,0);
    conv(nullptr, u4,  dw5,  db5, nullptr, recon,16,136,512,  3,136,512, 1,0,0, 2, 1,1);
}
